// S_decoder_12137577578919
// MI455X (gfx1250) — compile-verified, matrix operand order checked by reading, not run on hardware
//
#include <hip/hip_runtime.h>
#include <math.h>

typedef __attribute__((ext_vector_type(16))) __bf16    v16bf;
typedef __attribute__((ext_vector_type(2)))  __bf16    v2bf;
typedef __attribute__((ext_vector_type(8)))  float     v8f;
typedef __attribute__((ext_vector_type(8)))  unsigned  v8u;

#define NN 100000
#define EE 1600000
#define DD 64
#define HH 64
#define BN_EPS 1e-5f

// ---------------------------------------------------------------------------
// Init: deg = 1 (self-loop), zero BN stat accumulators
// ---------------------------------------------------------------------------
__global__ void init_kernel(float* deg, float* st1, float* st2) {
    int i = blockIdx.x * blockDim.x + threadIdx.x;
    if (i < NN) deg[i] = 1.0f;
    if (i < 128) st1[i] = 0.0f;
    if (i < 4)   st2[i] = 0.0f;
}

// deg[dst] += 1 per edge
__global__ void deg_kernel(const int* dst, float* deg) {
    int i = blockIdx.x * blockDim.x + threadIdx.x;
    int stride = gridDim.x * blockDim.x;
    for (int e = i; e < EE; e += stride)
        atomicAdd(&deg[dst[e]], 1.0f);
}

// deg -> rsqrt(deg) in place
__global__ void dinv_kernel(float* deg) {
    int i = blockIdx.x * blockDim.x + threadIdx.x;
    if (i < NN) deg[i] = rsqrtf(deg[i]);
}

// ---------------------------------------------------------------------------
// WMMA fragment layouts (wave32):
// A 16x32 bf16 (MxK): lane m = lane&15, half = lane>>4; VGPR v packs K = kb,kb+1
//   kb = k0 + (v>=4 ? 16 : 0) + half*8 + (v&3)*2
// B 32x16 bf16 (KxN): mirrored, lane n = lane&15
// C 16x16 f32: VGPR r -> row r + half*8, col = lane&15
// ---------------------------------------------------------------------------

// Pre-pack B fragments from fp32 W (64 x nvalid, row-major, leading dim ld)
// into bf16-pair u32 words in register layout, zero-padded beyond nvalid.
// out index = ((ntile*2 + kchunk)*32 + lane)*8 + v   -> lane reads 32B blocks.
__global__ void pack_B_kernel(const float* __restrict__ W, int ld, int nvalid,
                              int ntiles, unsigned* __restrict__ out) {
    int i = blockIdx.x * blockDim.x + threadIdx.x;
    int total = ntiles * 2 * 32 * 8;
    if (i >= total) return;
    int v    = i & 7;
    int lane = (i >> 3) & 31;
    int kc   = (i >> 8) & 1;
    int t    = i >> 9;
    int n    = t * 16 + (lane & 15);
    int half = (lane >> 4) & 1;
    int kb   = kc * 32 + ((v >= 4) ? 16 : 0) + half * 8 + (v & 3) * 2;
    float x0 = (n < nvalid) ? W[(size_t)kb * ld + n]       : 0.0f;
    float x1 = (n < nvalid) ? W[(size_t)(kb + 1) * ld + n] : 0.0f;
    v2bf p;
    p[0] = (__bf16)x0;
    p[1] = (__bf16)x1;
    out[i] = __builtin_bit_cast(unsigned, p);
}

// A fragment: two contiguous 32B runs per lane -> vectorizes to b128 loads
__device__ inline v16bf load_A_bf16(const float* X, int row0, int ldx, int k0, int lane) {
    const float* rp = X + (size_t)(row0 + (lane & 15)) * ldx;
    int half = (lane >> 4) & 1;
    v16bf a;
#pragma unroll
    for (int v = 0; v < 8; ++v) {
        int kb = k0 + ((v >= 4) ? 16 : 0) + half * 8 + (v & 3) * 2;
        a[2 * v]     = (__bf16)rp[kb];
        a[2 * v + 1] = (__bf16)rp[kb + 1];
    }
    return a;
}

__device__ inline v16bf load_B_packed(const unsigned* Bp, int tile, int kc, int lane) {
    v8u bu = *(const v8u*)(Bp + ((((tile * 2 + kc) * 32) + lane) << 3));
    return __builtin_bit_cast(v16bf, bu);
}

// H1 = X(N x 64) @ W(64 x 64); one wave per 16x16 tile, 4 n-tiles per block
__global__ void gemm1_wmma(const float* __restrict__ X, const unsigned* __restrict__ Bp,
                           float* __restrict__ H1) {
    int lane = threadIdx.x & 31;
    int wave = threadIdx.x >> 5;            // n-tile 0..3
    int row0 = blockIdx.x * 16;             // 6250 m-tiles, exact
    v8f c = {};
#pragma unroll
    for (int kc = 0; kc < 2; ++kc) {
        v16bf a = load_A_bf16(X, row0, DD, kc * 32, lane);
        v16bf b = load_B_packed(Bp, wave, kc, lane);
        c = __builtin_amdgcn_wmma_f32_16x16x32_bf16(false, a, false, b, (short)0, c, false, false);
    }
    int n = wave * 16 + (lane & 15);
    int rbase = row0 + ((lane >> 4) & 1) * 8;
#pragma unroll
    for (int r = 0; r < 8; ++r)
        H1[(size_t)(rbase + r) * HH + n] = c[r];
}

// H2 = X(N x 64) @ W2(64 x 2); B pre-packed with zero padding; 4 m-tiles/block
__global__ void gemm2_wmma(const float* __restrict__ X, const unsigned* __restrict__ Bp,
                           float* __restrict__ H2) {
    int lane = threadIdx.x & 31;
    int wave = threadIdx.x >> 5;
    int mtile = blockIdx.x * 4 + wave;
    if (mtile >= NN / 16) return;
    int row0 = mtile * 16;
    v8f c = {};
#pragma unroll
    for (int kc = 0; kc < 2; ++kc) {
        v16bf a = load_A_bf16(X, row0, HH, kc * 32, lane);
        v16bf b = load_B_packed(Bp, 0, kc, lane);
        c = __builtin_amdgcn_wmma_f32_16x16x32_bf16(false, a, false, b, (short)0, c, false, false);
    }
    int n = lane & 15;
    if (n < 2) {
        int rbase = row0 + ((lane >> 4) & 1) * 8;
#pragma unroll
        for (int r = 0; r < 8; ++r)
            H2[(size_t)(rbase + r) * 2 + n] = c[r];
    }
}

// ---------------------------------------------------------------------------
// A[i] = H[i] * dinv[row]^2 + bias[col]   (initializes aggregation buffer:
// self-loop contribution + post-aggregation bias, no pre-zeroing needed)
// ---------------------------------------------------------------------------
template <int NC>
__global__ void self_bias(const float* __restrict__ Hm, const float* __restrict__ dinv,
                          const float* __restrict__ bias, float* __restrict__ A) {
    int i = blockIdx.x * blockDim.x + threadIdx.x;
    int total = NN * NC;
    int stride = gridDim.x * blockDim.x;
    for (; i < total; i += stride) {
        int r = i / NC, c = i - r * NC;
        float di = dinv[r];
        A[i] = Hm[i] * di * di + bias[c];
    }
}

// one wave per edge, 32 lanes x float2 = 64 features; prefetch next src row
__global__ void edge_agg64(const int* __restrict__ src, const int* __restrict__ dst,
                           const float* __restrict__ dinv, const float* __restrict__ Hm,
                           float* __restrict__ A) {
    int lane = threadIdx.x & 31;
    int wave = (blockIdx.x * blockDim.x + threadIdx.x) >> 5;
    int nwaves = (gridDim.x * blockDim.x) >> 5;
    for (int e = wave; e < EE; e += nwaves) {
        int s = src[e], d = dst[e];
        if (e + nwaves < EE)
            __builtin_prefetch(&Hm[(size_t)src[e + nwaves] * HH], 0, 1);
        float w = dinv[s] * dinv[d];
        float2 h = ((const float2*)(Hm + (size_t)s * HH))[lane];
        float* ap = A + (size_t)d * HH + 2 * lane;
        atomicAdd(ap,     h.x * w);
        atomicAdd(ap + 1, h.y * w);
    }
}

// one thread per edge, 2 features
__global__ void edge_agg2(const int* __restrict__ src, const int* __restrict__ dst,
                          const float* __restrict__ dinv, const float* __restrict__ Hm,
                          float* __restrict__ A) {
    int i = blockIdx.x * blockDim.x + threadIdx.x;
    int stride = gridDim.x * blockDim.x;
    for (int e = i; e < EE; e += stride) {
        int s = src[e], d = dst[e];
        float w = dinv[s] * dinv[d];
        atomicAdd(&A[(size_t)d * 2],     Hm[(size_t)s * 2]     * w);
        atomicAdd(&A[(size_t)d * 2 + 1], Hm[(size_t)s * 2 + 1] * w);
    }
}

// ---------------------------------------------------------------------------
// BatchNorm stats: st[c] = sum, st[NC + c] = sumsq (LDS partials -> 1 global
// atomic per column per block)
// ---------------------------------------------------------------------------
__global__ void bn_stats64(const float* __restrict__ X, float* __restrict__ st) {
    __shared__ float ssum[64], ssq[64];
    int t = threadIdx.x;
    if (t < 64) { ssum[t] = 0.0f; ssq[t] = 0.0f; }
    __syncthreads();
    int col = t & 63;
    int rowsPerBlk = blockDim.x >> 6;
    int row = blockIdx.x * rowsPerBlk + (t >> 6);
    int rstride = gridDim.x * rowsPerBlk;
    float lsum = 0.0f, lsq = 0.0f;
    for (; row < NN; row += rstride) {
        float x = X[(size_t)row * 64 + col];
        lsum += x; lsq += x * x;
    }
    atomicAdd(&ssum[col], lsum);
    atomicAdd(&ssq[col], lsq);
    __syncthreads();
    if (t < 64) {
        atomicAdd(&st[t],      ssum[t]);
        atomicAdd(&st[64 + t], ssq[t]);
    }
}

__global__ void bn_stats2(const float* __restrict__ X, float* __restrict__ st) {
    __shared__ float acc[4];
    int t = threadIdx.x;
    if (t < 4) acc[t] = 0.0f;
    __syncthreads();
    float s0 = 0, s1 = 0, q0 = 0, q1 = 0;
    int i = blockIdx.x * blockDim.x + t;
    int stride = gridDim.x * blockDim.x;
    for (int r = i; r < NN; r += stride) {
        float x0 = X[(size_t)r * 2], x1 = X[(size_t)r * 2 + 1];
        s0 += x0; q0 += x0 * x0; s1 += x1; q1 += x1 * x1;
    }
    atomicAdd(&acc[0], s0); atomicAdd(&acc[1], s1);
    atomicAdd(&acc[2], q0); atomicAdd(&acc[3], q1);
    __syncthreads();
    if (t < 4) atomicAdd(&st[t], acc[t]);
}

// BN apply (+ optional ReLU) in place
template <int NC, bool RELU>
__global__ void bn_apply(float* __restrict__ X, const float* __restrict__ st,
                         const float* __restrict__ g, const float* __restrict__ beta) {
    int i = blockIdx.x * blockDim.x + threadIdx.x;
    int total = NN * NC;
    int stride = gridDim.x * blockDim.x;
    const float invN = 1.0f / (float)NN;
    for (; i < total; i += stride) {
        int c = i % NC;
        float mean = st[c] * invN;
        float var  = st[NC + c] * invN - mean * mean;
        float y = g[c] * (X[i] - mean) * rsqrtf(var + BN_EPS) + beta[c];
        if (RELU) y = fmaxf(y, 0.0f);
        X[i] = y;
    }
}

// ---------------------------------------------------------------------------
// Final: BN(2) + sigmoid -> out[0..2N), softmax(dim=1) -> out[2N..4N)
// ---------------------------------------------------------------------------
__global__ void finalize_kernel(const float* __restrict__ A2, const float* __restrict__ st,
                                const float* __restrict__ g2, const float* __restrict__ beta2,
                                float* __restrict__ out) {
    int i = blockIdx.x * blockDim.x + threadIdx.x;
    int stride = gridDim.x * blockDim.x;
    const float invN = 1.0f / (float)NN;
    float m0 = st[0] * invN, m1 = st[1] * invN;
    float v0 = st[2] * invN - m0 * m0;
    float v1 = st[3] * invN - m1 * m1;
    float sc0 = g2[0] * rsqrtf(v0 + BN_EPS);
    float sc1 = g2[1] * rsqrtf(v1 + BN_EPS);
    for (int r = i; r < NN; r += stride) {
        float s0 = sc0 * (A2[(size_t)r * 2]     - m0) + beta2[0];
        float s1 = sc1 * (A2[(size_t)r * 2 + 1] - m1) + beta2[1];
        out[(size_t)r * 2]     = 1.0f / (1.0f + __expf(-s0));
        out[(size_t)r * 2 + 1] = 1.0f / (1.0f + __expf(-s1));
        float mx = fmaxf(s0, s1);
        float e0 = __expf(s0 - mx), e1 = __expf(s1 - mx);
        float inv = 1.0f / (e0 + e1);
        out[(size_t)(NN + r) * 2]     = e0 * inv;
        out[(size_t)(NN + r) * 2 + 1] = e1 * inv;
    }
}

// ---------------------------------------------------------------------------
extern "C" void kernel_launch(void* const* d_in, const int* in_sizes, int n_in,
                              void* d_out, int out_size, void* d_ws, size_t ws_size,
                              hipStream_t stream) {
    const int*   edge = (const int*)d_in[0];
    const int*   esrc = edge;          // edge_index[0]
    const int*   edst = edge + EE;     // edge_index[1]
    const float* uS   = (const float*)d_in[1];
    const float* w1   = (const float*)d_in[2];
    const float* b1   = (const float*)d_in[3];
    const float* g1   = (const float*)d_in[4];
    const float* be1  = (const float*)d_in[5];
    const float* w2   = (const float*)d_in[6];
    const float* b2   = (const float*)d_in[7];
    const float* g2   = (const float*)d_in[8];
    const float* be2  = (const float*)d_in[9];
    float* out = (float*)d_out;

    float* ws    = (float*)d_ws;
    float* deg   = ws;                         // N   (deg, then dinv in place)
    float* h1    = deg + NN;                   // N*64
    float* a1    = h1 + (size_t)NN * HH;       // N*64
    float* h2    = a1 + (size_t)NN * HH;       // N*2
    float* a2    = h2 + (size_t)NN * 2;        // N*2
    unsigned* bp1 = (unsigned*)(a2 + (size_t)NN * 2);  // 2048 u32 (32B-aligned)
    unsigned* bp2 = bp1 + 2048;                        // 512 u32
    float* st1   = (float*)(bp2 + 512);        // 128
    float* st2   = st1 + 128;                  // 4

    // degree / normalization + weight fragment packing
    init_kernel<<<(NN + 255) / 256, 256, 0, stream>>>(deg, st1, st2);
    deg_kernel<<<2048, 256, 0, stream>>>(edst, deg);
    dinv_kernel<<<(NN + 255) / 256, 256, 0, stream>>>(deg);
    pack_B_kernel<<<8, 256, 0, stream>>>(w1, HH, HH, 4, bp1);
    pack_B_kernel<<<2, 256, 0, stream>>>(w2, 2, 2, 1, bp2);

    // layer 1: transform (WMMA) -> self-loop+bias init -> edge scatter -> BN+ReLU
    gemm1_wmma<<<NN / 16, 128, 0, stream>>>(uS, bp1, h1);
    self_bias<HH><<<4096, 256, 0, stream>>>(h1, deg, b1, a1);
    edge_agg64<<<4096, 256, 0, stream>>>(esrc, edst, deg, h1, a1);
    bn_stats64<<<512, 256, 0, stream>>>(a1, st1);
    bn_apply<HH, true><<<4096, 256, 0, stream>>>(a1, st1, g1, be1);

    // layer 2
    gemm2_wmma<<<(NN / 16 + 3) / 4, 128, 0, stream>>>(a1, bp2, h2);
    self_bias<2><<<1024, 256, 0, stream>>>(h2, deg, b2, a2);
    edge_agg2<<<2048, 256, 0, stream>>>(esrc, edst, deg, h2, a2);
    bn_stats2<<<512, 256, 0, stream>>>(a2, st2);

    // BN(2) + sigmoid + softmax
    finalize_kernel<<<(NN + 255) / 256, 256, 0, stream>>>(a2, st2, g2, be2, out);
}
